// CrossAttention_56753697849591
// MI455X (gfx1250) — matrix operand order repaired and checked by reading, not compiled, not run on hardware
//
#include <hip/hip_runtime.h>

// ---------- types matching the gfx1250 WMMA builtin signatures ----------
typedef __attribute__((ext_vector_type(16))) __bf16 v16bf;
typedef __attribute__((ext_vector_type(8)))  float  v8f;

union Frag {
    v16bf    v;
    unsigned u[8];
};

__device__ inline unsigned short f2bf(float f) {
    union { float f; unsigned u; } x;
    x.f = f;
    unsigned r = x.u + 0x7fffu + ((x.u >> 16) & 1u);   // RNE
    return (unsigned short)(r >> 16);
}
__device__ inline unsigned packbf(float a, float b) {
    return (unsigned)f2bf(a) | ((unsigned)f2bf(b) << 16);
}
__device__ inline v8f vzero() {
    v8f z;
#pragma unroll
    for (int i = 0; i < 8; ++i) z[i] = 0.f;
    return z;
}

// ============================================================
// Staging helper: load one 128x32 A tile and one 32x128 B tile
// into pair-packed bf16 LDS buffers; prefetch the next chunk.
// ============================================================
template <bool A_BF16>
__device__ __forceinline__ void stage_AB(
    unsigned* __restrict__ As, unsigned* __restrict__ Bs,
    const void* __restrict__ Ap, const float* __restrict__ Bw,
    long blockM, int blockN, int kb, int M, int K, int N, int tid)
{
    // A tile (128 rows x 16 bf16-pairs)
    for (int i = tid; i < 2048; i += 256) {
        int row = i >> 4, k2 = i & 15;
        long gr = blockM + row;
        unsigned val = 0u;
        if (gr < M) {
            if (A_BF16) {
                const unsigned* Au = (const unsigned*)Ap;
                long idx = ((gr * (long)K + kb) >> 1) + k2;
                val = Au[idx];
                __builtin_prefetch(&Au[idx + 16], 0, 1);      // next K-chunk
            } else {
                const float* Af = (const float*)Ap;
                long base = gr * (long)K + kb + 2 * k2;
                val = packbf(Af[base], Af[base + 1]);
                __builtin_prefetch(&Af[base + 32], 0, 1);     // next K-chunk
            }
        }
        As[row * 16 + k2] = val;
    }
    // B tile (16 k-pairs x 128 cols), weights always f32
    for (int i = tid; i < 2048; i += 256) {
        int k2 = i >> 7, n = i & 127;
        int gc = blockN + n;
        float f0 = 0.f, f1 = 0.f;
        if (gc < N) {
            f0 = Bw[(long)(kb + 2 * k2) * N + gc];
            f1 = Bw[(long)(kb + 2 * k2 + 1) * N + gc];
        }
        Bs[k2 * 128 + n] = packbf(f0, f1);
    }
}

// ============================================================
// Tiled bf16 WMMA GEMM: C[M,N] = A[M,K] * B[K,N] (+bias)
// Block tile 128x128, K-chunk 32, 256 threads = 8 waves (4x2),
// double-buffered LDS (one barrier per K-chunk).
// ============================================================
template <bool A_BF16, bool BIAS, bool OUT_BF16>
__global__ __launch_bounds__(256) void gemm_bf16_wmma(
    const void* __restrict__ Ap, const float* __restrict__ Bw,
    const float* __restrict__ bias, void* __restrict__ Cp,
    int M, int K, int N)
{
    __shared__ unsigned As[2][128 * 16];   // [buf][row][k2]
    __shared__ unsigned Bs[2][16 * 128];   // [buf][k2][col]

    const int tid  = threadIdx.x;
    const int lane = tid & 31;
    const int wv   = tid >> 5;
    const int wm   = wv & 3;        // 4 wave rows  (32 rows each)
    const int wn   = wv >> 2;       // 2 wave cols  (64 cols each)
    const int n0   = lane & 15;
    const int half = lane >> 4;
    const int  blockN = blockIdx.x * 128;
    const long blockM = (long)blockIdx.y * 128;

    v8f acc[2][4];
#pragma unroll
    for (int mt = 0; mt < 2; ++mt)
#pragma unroll
        for (int nt = 0; nt < 4; ++nt) acc[mt][nt] = vzero();

    const int nK = K >> 5;
    stage_AB<A_BF16>(As[0], Bs[0], Ap, Bw, blockM, blockN, 0, M, K, N, tid);
    __syncthreads();

    for (int kc = 0; kc < nK; ++kc) {
        const int cur = kc & 1;
        unsigned* Ac = As[cur];
        unsigned* Bc = Bs[cur];
        if (kc + 1 < nK)
            stage_AB<A_BF16>(As[cur ^ 1], Bs[cur ^ 1], Ap, Bw, blockM, blockN,
                             (kc + 1) << 5, M, K, N, tid);

        Frag a[2], b[4];
#pragma unroll
        for (int mt = 0; mt < 2; ++mt) {
            int row = wm * 32 + mt * 16 + n0;
#pragma unroll
            for (int r = 0; r < 8; ++r) {
                int k2 = half * 8 + r;
                a[mt].u[r] = Ac[row * 16 + k2];
            }
        }
#pragma unroll
        for (int nt = 0; nt < 4; ++nt) {
            int col = wn * 64 + nt * 16 + n0;
#pragma unroll
            for (int r = 0; r < 8; ++r) {
                int k2 = half * 8 + r;
                b[nt].u[r] = Bc[k2 * 128 + col];
            }
        }
#pragma unroll
        for (int mt = 0; mt < 2; ++mt)
#pragma unroll
            for (int nt = 0; nt < 4; ++nt)
                acc[mt][nt] = __builtin_amdgcn_wmma_f32_16x16x32_bf16(
                    false, a[mt].v, false, b[nt].v, (short)0, acc[mt][nt],
                    false, false);
        __syncthreads();   // protects both: readers of cur, writers of cur^1
    }

    // ---- writeback: C layout vgpr r -> M=r+8*half, lane -> N=n0 ----
#pragma unroll
    for (int mt = 0; mt < 2; ++mt) {
#pragma unroll
        for (int nt = 0; nt < 4; ++nt) {
            int col = blockN + wn * 64 + nt * 16 + n0;
            float bv = 0.f;
            if (BIAS) bv = (col < N) ? bias[col] : 0.f;
#pragma unroll
            for (int r = 0; r < 8; ++r) {
                long row = blockM + wm * 32 + mt * 16 + r + half * 8;
                if (row < M && col < N) {
                    float vv = acc[mt][nt][r] + bv;
                    if (OUT_BF16)
                        ((unsigned short*)Cp)[row * (long)N + col] = f2bf(vv);
                    else
                        ((float*)Cp)[row * (long)N + col] = vv;
                }
            }
        }
    }
}

// ============================================================
// Fused attention for one (b, h, 64 q-rows) block.
// 128 threads = 4 waves; each wave owns 16 q rows.
// Skv=77 padded to 96, D=80 padded to 96 (zeros staged in LDS).
// ============================================================
__global__ __launch_bounds__(128) void attn_flash(
    const unsigned short* __restrict__ Q,   // bf16 [B*4096, 640]
    const unsigned short* __restrict__ Kp,  // bf16 [B*77,   640]
    const unsigned short* __restrict__ Vp,  // bf16 [B*77,   640]
    unsigned short* __restrict__ O)         // bf16 [B*4096, 640]
{
    __shared__ unsigned smem[13824];
    unsigned* kt = smem;          // K^T pairs: [d2 0..47][kv 0..79]
    unsigned* vt = smem + 3840;   // V  pairs:  [kv2 0..47][d 0..79]
    unsigned* qt = smem + 7680;   // Q  pairs:  [row 0..63][d2 0..47]
    unsigned* pt = smem + 10752;  // P  pairs:  [wave][m 0..15][kv2 0..47]

    const int tid  = threadIdx.x;
    const int lane = tid & 31;
    const int wv   = tid >> 5;
    const int n0   = lane & 15;
    const int half = lane >> 4;
    const int b     = blockIdx.y >> 3;
    const int h     = blockIdx.y & 7;
    const int qBase = blockIdx.x * 64;

    // zero all LDS (provides the zero padding everywhere)
    for (int i = tid; i < 13824; i += 128) smem[i] = 0u;
    __syncthreads();

    // ---- stage Q (64 x 80 valid, pairs contiguous in global) ----
    for (int i = tid; i < 2560; i += 128) {            // 64*40
        int row = i / 40, d2 = i % 40;
        long gro = (long)(b * 4096 + qBase + row) * 640 + h * 80;
        qt[row * 48 + d2] = ((const unsigned*)(Q + gro))[d2];
    }
    // ---- stage K^T (pairs over d are contiguous in K rows) ----
    for (int i = tid; i < 3200; i += 128) {            // 40*80
        int d2 = i / 80, kv = i % 80;
        if (kv < 77) {
            long gro = (long)(b * 77 + kv) * 640 + h * 80;
            kt[d2 * 80 + kv] = ((const unsigned*)(Kp + gro))[d2];
        }
    }
    // ---- stage V (pairs over kv: two strided u16 loads) ----
    for (int i = tid; i < 3120; i += 128) {            // 39*80
        int kv2 = i / 80, d = i % 80;
        int kv = kv2 * 2;
        unsigned v0 = Vp[(long)(b * 77 + kv) * 640 + h * 80 + d];
        unsigned v1 = (kv + 1 < 77)
                          ? (unsigned)Vp[(long)(b * 77 + kv + 1) * 640 + h * 80 + d]
                          : 0u;
        vt[kv2 * 80 + d] = v0 | (v1 << 16);
    }
    __syncthreads();

    // ---- scores S = Q K^T  (16 x 80 per wave), K-chunks of 32 over D=96 ----
    v8f s[5];
#pragma unroll
    for (int j = 0; j < 5; ++j) s[j] = vzero();
#pragma unroll
    for (int kc = 0; kc < 3; ++kc) {
        Frag a;
        int row = wv * 16 + n0;
#pragma unroll
        for (int r = 0; r < 8; ++r) {
            int k2 = kc * 16 + half * 8 + r;
            a.u[r] = qt[row * 48 + k2];
        }
#pragma unroll
        for (int j = 0; j < 5; ++j) {
            Frag bb;
            int kv = j * 16 + n0;
#pragma unroll
            for (int r = 0; r < 8; ++r) {
                int k2 = kc * 16 + half * 8 + r;
                bb.u[r] = kt[k2 * 80 + kv];
            }
            s[j] = __builtin_amdgcn_wmma_f32_16x16x32_bf16(
                false, a.v, false, bb.v, (short)0, s[j], false, false);
        }
    }

    // ---- row softmax (rows live in 16-lane halves; xor 1,2,4,8) ----
    const float scale = 0.11180339887498948f;  // 80^-0.5
    unsigned short* pw = (unsigned short*)(pt + wv * 768);
#pragma unroll
    for (int r = 0; r < 8; ++r) {
        int m = r + half * 8;
        float v[5];
        float mx = -1e30f;
#pragma unroll
        for (int j = 0; j < 5; ++j) {
            float x = s[j][r] * scale;
            if (j == 4 && n0 >= 13) x = -1e30f;  // mask kv >= 77
            v[j] = x;
            mx = fmaxf(mx, x);
        }
        for (int off = 1; off <= 8; off <<= 1)
            mx = fmaxf(mx, __shfl_xor(mx, off, 32));
        float sum = 0.f;
#pragma unroll
        for (int j = 0; j < 5; ++j) {
            float p = exp2f((v[j] - mx) * 1.4426950408889634f);
            v[j] = p;
            sum += p;
        }
        for (int off = 1; off <= 8; off <<= 1)
            sum += __shfl_xor(sum, off, 32);
        float rinv = 1.f / sum;
#pragma unroll
        for (int j = 0; j < 5; ++j)
            pw[m * 96 + j * 16 + n0] = f2bf(v[j] * rinv);
    }

    // ---- O = P V  (16 x 80 per wave), K-chunks of 32 over Skv=96 ----
    v8f o[5];
#pragma unroll
    for (int j = 0; j < 5; ++j) o[j] = vzero();
#pragma unroll
    for (int kc = 0; kc < 3; ++kc) {
        Frag a;
#pragma unroll
        for (int r = 0; r < 8; ++r) {
            int k2 = kc * 16 + half * 8 + r;
            a.u[r] = pt[wv * 768 + n0 * 48 + k2];
        }
#pragma unroll
        for (int j = 0; j < 5; ++j) {
            Frag bb;
            int d = j * 16 + n0;
#pragma unroll
            for (int r = 0; r < 8; ++r) {
                int k2 = kc * 16 + half * 8 + r;
                bb.u[r] = vt[k2 * 80 + d];
            }
            o[j] = __builtin_amdgcn_wmma_f32_16x16x32_bf16(
                false, a.v, false, bb.v, (short)0, o[j], false, false);
        }
    }

    // ---- store attention output as bf16 [B*4096, 640] ----
#pragma unroll
    for (int j = 0; j < 5; ++j) {
#pragma unroll
        for (int r = 0; r < 8; ++r) {
            int m = r + half * 8;
            long row = (long)(b * 4096 + qBase + wv * 16 + m);
            O[row * 640 + h * 80 + j * 16 + n0] = f2bf(o[j][r]);
        }
    }
}

// ============================================================
// Launch: Q proj -> K proj -> V proj -> attention -> out proj
// ============================================================
extern "C" void kernel_launch(void* const* d_in, const int* in_sizes, int n_in,
                              void* d_out, int out_size, void* d_ws, size_t ws_size,
                              hipStream_t stream) {
    const float* x    = (const float*)d_in[0];  // [16,4096,640]
    const float* ctx  = (const float*)d_in[1];  // [16,77,768]
    const float* Wq   = (const float*)d_in[2];  // [640,640]
    const float* Wk   = (const float*)d_in[3];  // [768,640]
    const float* Wv   = (const float*)d_in[4];  // [768,640]
    const float* Wout = (const float*)d_in[5];  // [640,640]
    const float* bout = (const float*)d_in[6];  // [640]
    float* out = (float*)d_out;

    char* ws = (char*)d_ws;
    unsigned short* Qb = (unsigned short*)(ws);              // 65536*640 bf16
    unsigned short* Kb = (unsigned short*)(ws + 83886080);   // 1232*640 bf16
    unsigned short* Vb = (unsigned short*)(ws + 85463040);   // 1232*640 bf16
    unsigned short* Ab = (unsigned short*)(ws + 87040000);   // 65536*640 bf16

    dim3 blk(256);
    // Q = x @ Wq            (M=65536, K=640, N=640)
    gemm_bf16_wmma<false, false, true><<<dim3(5, 512), blk, 0, stream>>>(
        x, Wq, nullptr, Qb, 65536, 640, 640);
    // K = ctx @ Wk          (M=1232, K=768, N=640)
    gemm_bf16_wmma<false, false, true><<<dim3(5, 10), blk, 0, stream>>>(
        ctx, Wk, nullptr, Kb, 1232, 768, 640);
    // V = ctx @ Wv
    gemm_bf16_wmma<false, false, true><<<dim3(5, 10), blk, 0, stream>>>(
        ctx, Wv, nullptr, Vb, 1232, 768, 640);
    // attention: grid (Sq/64, B*H)
    attn_flash<<<dim3(64, 128), dim3(128), 0, stream>>>(Qb, Kb, Vb, Ab);
    // out = attn @ Wout + bout  (f32 output)
    gemm_bf16_wmma<true, true, false><<<dim3(5, 512), blk, 0, stream>>>(
        Ab, Wout, bout, out, 65536, 640, 640);
}
